// EncoderLayer_78735340471038
// MI455X (gfx1250) — compile-verified, matrix operand order checked by reading, not run on hardware
//
#include <hip/hip_runtime.h>
#include <math.h>

// Problem constants (from reference)
#define B_   8
#define S_   2048
#define D_   1024
#define H_   16
#define HD_  64
#define E_   8
#define DFF_ 4096
#define N_   (B_ * S_)          // 16384 tokens
#define TD_  (3 * D_)           // 3072
#define CAP_ 2560               // int(1.25 * N / E)

typedef __attribute__((ext_vector_type(16))) __bf16 v16bf;
typedef __attribute__((ext_vector_type(8)))  float  v8f;
typedef __attribute__((ext_vector_type(4)))  int    v4i_t;

typedef __attribute__((address_space(1))) v4i_t glb_v4i;
typedef __attribute__((address_space(3))) v4i_t lds_v4i;

// ---------------------------------------------------------------------------
// 16-byte async global->LDS copy (CDNA5 GLOBAL_LOAD_ASYNC_TO_LDS_B128,
// tracked by ASYNCcnt), with inline-asm fallback.
// ---------------------------------------------------------------------------
__device__ __forceinline__ void async_copy16(const void* g, void* l) {
#if __has_builtin(__builtin_amdgcn_global_load_async_to_lds_b128)
  __builtin_amdgcn_global_load_async_to_lds_b128(
      (glb_v4i*)g, (lds_v4i*)l, 0, 0);
#else
  asm volatile("global_load_async_to_lds_b128 %0, %1, off"
               :: "v"((unsigned)(unsigned long long)(size_t)l),
                  "v"((unsigned long long)(size_t)g)
               : "memory");
#endif
}

template<int N>
__device__ __forceinline__ void wait_async() {
#if __has_builtin(__builtin_amdgcn_s_wait_asynccnt)
  __builtin_amdgcn_s_wait_asynccnt(N);
#else
  asm volatile("s_wait_asynccnt %0" :: "i"(N) : "memory");
#endif
}

// ---------------------------------------------------------------------------
// WMMA 16-bit A/B fragment loader from an LDS tile stored as [row][k] (bf16).
// ISA 7.12.2: lane m = lane%16, group g = lane>>4, VGPR r holds K pair
// k = (r<4 ? 2r : 16+2(r-4)) + 8g.  B uses the mirrored layout from [n][k].
// ---------------------------------------------------------------------------
template<int LD>
__device__ __forceinline__ v16bf load_frag(const __bf16* base, int lane) {
  const int m = lane & 15, g = lane >> 4;
  v16bf f;
#pragma unroll
  for (int r = 0; r < 8; ++r) {
    const int k = ((r < 4) ? (2 * r) : (16 + 2 * (r - 4))) + 8 * g;
    f[2 * r]     = base[m * LD + k];
    f[2 * r + 1] = base[m * LD + k + 1];
  }
  return f;
}

// ---------------------------------------------------------------------------
// NT WMMA GEMM, bf16 in / fp32-or-bf16 out, f32 accumulate.
//   C[M,Ncols] = A[M,K] * B^T      (B given as [Ncols,K] row-major)
// Double-buffered async LDS staging: tile t+1 copies overlap tile t compute.
// Block: 256 threads = 8 waves; block tile 128x128; wave tile 64x32 (4x2
// wmma). Batched over blockIdx.z with element strides.
// ---------------------------------------------------------------------------
__global__ __launch_bounds__(256) void gemm_wmma_nt(
    const __bf16* __restrict__ A, const __bf16* __restrict__ Bm,
    void* __restrict__ Cv, const float* __restrict__ bias,
    const float* __restrict__ resid,
    int M, int Ncols, int K,
    long sA, long sB, long sC, long sBias, int relu, int c_bf16)
{
  __shared__ __align__(16) __bf16 As[2][128][40];
  __shared__ __align__(16) __bf16 Bs[2][128][40];

  const int tid  = threadIdx.x;
  const int lane = tid & 31, wid = tid >> 5;
  const int wm = wid & 1;        // 2 waves along M (64 rows each)
  const int wn = wid >> 1;       // 4 waves along N (32 cols each)
  const int bm = blockIdx.y * 128, bn = blockIdx.x * 128;

  A  += (long)blockIdx.z * sA;
  Bm += (long)blockIdx.z * sB;
  const float* bp = bias ? (bias + (long)blockIdx.z * sBias) : nullptr;
  const long cbase = (long)blockIdx.z * sC;

  v8f acc[4][2] = {};

  // async copy mapping: each thread moves 2x16B for A and 2x16B for B per tile
  const int r0 = tid >> 2;             // 0..63
  const int ch = (tid & 3) * 16;       // byte offset within 64B row chunk

  auto issue_tile = [&](int k0, int buf) {
#pragma unroll
    for (int i = 0; i < 2; ++i) {
      const int r = r0 + 64 * i;
      async_copy16((const char*)A  + ((long)(bm + r) * K + k0) * 2 + ch,
                   (char*)&As[buf][r][0] + ch);
      async_copy16((const char*)Bm + ((long)(bn + r) * K + k0) * 2 + ch,
                   (char*)&Bs[buf][r][0] + ch);
    }
  };

  const int T = K / 32;
  issue_tile(0, 0);

  for (int t = 0; t < T; ++t) {
    const int buf = t & 1;
    if (t + 1 < T) {
      issue_tile((t + 1) * 32, (t + 1) & 1);  // prefetch next tile
      wait_async<4>();   // in-order: current tile's 4 copies are complete
    } else {
      wait_async<0>();
    }
    __syncthreads();     // all waves' current tile visible in LDS

    v16bf bf[2];
#pragma unroll
    for (int nt = 0; nt < 2; ++nt)
      bf[nt] = load_frag<40>(&Bs[buf][wn * 32 + nt * 16][0], lane);
#pragma unroll
    for (int mt = 0; mt < 4; ++mt) {
      const v16bf af = load_frag<40>(&As[buf][wm * 64 + mt * 16][0], lane);
#pragma unroll
      for (int nt = 0; nt < 2; ++nt)
        acc[mt][nt] = __builtin_amdgcn_wmma_f32_16x16x32_bf16(
            false, af, false, bf[nt], (short)0, acc[mt][nt], false, false);
    }
    // No trailing barrier: buffer `buf` is next written by the prefetch at iteration
    // t+1, issued before the t+1 barrier, so it can overlap other waves'
    // reads of `buf` here.
  }

  // Epilogue. C layout: row = r + 8*(lane>>4), col = lane&15 within 16x16.
  const int ni = lane & 15, g = lane >> 4;
#pragma unroll
  for (int mt = 0; mt < 4; ++mt)
#pragma unroll
    for (int nt = 0; nt < 2; ++nt) {
      const int col = bn + wn * 32 + nt * 16 + ni;
      const float bv = bp ? bp[col] : 0.0f;
#pragma unroll
      for (int r = 0; r < 8; ++r) {
        const int row = bm + wm * 64 + mt * 16 + r + 8 * g;
        float v = acc[mt][nt][r] + bv;
        if (relu) v = fmaxf(v, 0.0f);
        if (resid) v += resid[(long)row * Ncols + col];
        if (c_bf16) ((__bf16*)Cv)[cbase + (long)row * Ncols + col] = (__bf16)v;
        else        ((float*)Cv)[cbase + (long)row * Ncols + col] = v;
      }
    }
}

// ---------------------------------------------------------------------------
// Flash attention: grid = (S/128, B*H), block = 256 (8 waves, 16 q-rows each).
// qkv (fp32) layout: [N, 3D]; q at col h*64, k at D + h*64, v at 2D + h*64.
// Output ctx is bf16 (feeds the out-projection GEMM directly).
// ---------------------------------------------------------------------------
__global__ __launch_bounds__(256) void flash_attn(
    const float* __restrict__ qkv, const unsigned char* __restrict__ pmask,
    __bf16* __restrict__ ctx)
{
  __shared__ __align__(16) __bf16 Qs[128][72];
  __shared__ __align__(16) __bf16 Ks[128][72];
  __shared__ __align__(16) __bf16 Vt[64][136];    // [hd][kv] (B layout [n][k])
  __shared__ __align__(16) __bf16 Ps[128][136];   // [q-row][kv] (A layout)

  const int tid = threadIdx.x, lane = tid & 31, w = tid >> 5;
  const int b = blockIdx.y >> 4, h = blockIdx.y & 15;
  const int q0 = blockIdx.x * 128;
  const long rowbase = (long)b * S_;
  const int qcol = h * HD_;

  // Load Q tile (scaled by 1/sqrt(hd) = 0.125, folded into Q).
#pragma unroll
  for (int i = 0; i < 8; ++i) {
    const int j = tid + 256 * i;
    const int r = j >> 4, c = (j & 15) * 4;
    const float4 v = *(const float4*)(qkv + (rowbase + q0 + r) * TD_ + qcol + c);
    Qs[r][c + 0] = (__bf16)(v.x * 0.125f); Qs[r][c + 1] = (__bf16)(v.y * 0.125f);
    Qs[r][c + 2] = (__bf16)(v.z * 0.125f); Qs[r][c + 3] = (__bf16)(v.w * 0.125f);
  }
  __syncthreads();
  v16bf qf[2];
  qf[0] = load_frag<72>(&Qs[w * 16][0], lane);
  qf[1] = load_frag<72>(&Qs[w * 16][32], lane);

  const int g = lane >> 4, ni = lane & 15;
  float mrow[8], lrow[8];
  v8f oacc[4] = {};
#pragma unroll
  for (int r = 0; r < 8; ++r) { mrow[r] = -1e30f; lrow[r] = 0.0f; }

  for (int kt = 0; kt < S_ / 128; ++kt) {
    const int kv0 = kt * 128;
    __syncthreads();   // previous PV reads done before overwriting Ks/Vt
#pragma unroll
    for (int i = 0; i < 8; ++i) {
      const int j = tid + 256 * i;
      const int r = j >> 4, c = (j & 15) * 4;
      const float4 kv = *(const float4*)(qkv + (rowbase + kv0 + r) * TD_ + D_ + qcol + c);
      Ks[r][c + 0] = (__bf16)kv.x; Ks[r][c + 1] = (__bf16)kv.y;
      Ks[r][c + 2] = (__bf16)kv.z; Ks[r][c + 3] = (__bf16)kv.w;
      const float4 vv = *(const float4*)(qkv + (rowbase + kv0 + r) * TD_ + 2 * D_ + qcol + c);
      Vt[c + 0][r] = (__bf16)vv.x; Vt[c + 1][r] = (__bf16)vv.y;
      Vt[c + 2][r] = (__bf16)vv.z; Vt[c + 3][r] = (__bf16)vv.w;
    }
    __syncthreads();

    // scores: 16x128 per wave = 8 n-tiles x 2 k-steps
    v8f s[8] = {};
#pragma unroll
    for (int nt = 0; nt < 8; ++nt) {
      s[nt] = __builtin_amdgcn_wmma_f32_16x16x32_bf16(false, qf[0], false,
                load_frag<72>(&Ks[nt * 16][0], lane), (short)0, s[nt], false, false);
      s[nt] = __builtin_amdgcn_wmma_f32_16x16x32_bf16(false, qf[1], false,
                load_frag<72>(&Ks[nt * 16][32], lane), (short)0, s[nt], false, false);
    }
    // key-padding mask (column = ni within tile)
#pragma unroll
    for (int nt = 0; nt < 8; ++nt) {
      const float mneg = pmask[b * S_ + kv0 + nt * 16 + ni] ? -1e9f : 0.0f;
#pragma unroll
      for (int r = 0; r < 8; ++r) s[nt][r] += mneg;
    }
    // online softmax: rows live in (r, g); reduce over 16 lanes of the group
    float tmax[8], rsum[8], mnew[8], alpha[8];
#pragma unroll
    for (int r = 0; r < 8; ++r) {
      tmax[r] = s[0][r];
#pragma unroll
      for (int nt = 1; nt < 8; ++nt) tmax[r] = fmaxf(tmax[r], s[nt][r]);
    }
#pragma unroll
    for (int off = 1; off < 16; off <<= 1)
#pragma unroll
      for (int r = 0; r < 8; ++r)
        tmax[r] = fmaxf(tmax[r], __shfl_xor(tmax[r], off, 32));
#pragma unroll
    for (int r = 0; r < 8; ++r) {
      mnew[r] = fmaxf(mrow[r], tmax[r]);
      alpha[r] = __expf(mrow[r] - mnew[r]);
      rsum[r] = 0.0f;
    }
#pragma unroll
    for (int nt = 0; nt < 8; ++nt)
#pragma unroll
      for (int r = 0; r < 8; ++r) {
        const float p = __expf(s[nt][r] - mnew[r]);
        rsum[r] += p;
        Ps[w * 16 + r + 8 * g][nt * 16 + ni] = (__bf16)p;
      }
#pragma unroll
    for (int off = 1; off < 16; off <<= 1)
#pragma unroll
      for (int r = 0; r < 8; ++r) rsum[r] += __shfl_xor(rsum[r], off, 32);
#pragma unroll
    for (int r = 0; r < 8; ++r) { lrow[r] = lrow[r] * alpha[r] + rsum[r]; mrow[r] = mnew[r]; }
#pragma unroll
    for (int vt = 0; vt < 4; ++vt)
#pragma unroll
      for (int r = 0; r < 8; ++r) oacc[vt][r] *= alpha[r];
    __syncthreads();   // Ps / Vt visible

    // P (16x128) * V (128x64): 4 n-tiles x 4 k-steps
#pragma unroll
    for (int vt = 0; vt < 4; ++vt)
#pragma unroll
      for (int ks = 0; ks < 4; ++ks)
        oacc[vt] = __builtin_amdgcn_wmma_f32_16x16x32_bf16(false,
            load_frag<136>(&Ps[w * 16][ks * 32], lane), false,
            load_frag<136>(&Vt[vt * 16][ks * 32], lane), (short)0, oacc[vt], false, false);
  }

#pragma unroll
  for (int vt = 0; vt < 4; ++vt)
#pragma unroll
    for (int r = 0; r < 8; ++r) {
      const long row = rowbase + q0 + w * 16 + r + 8 * g;
      ctx[row * D_ + qcol + vt * 16 + ni] =
          (__bf16)(oacc[vt][r] / fmaxf(lrow[r], 1e-20f));
    }
}

// ---------------------------------------------------------------------------
// LayerNorm over D=1024; one block (256 threads) per token; bf16 output.
// ---------------------------------------------------------------------------
__global__ __launch_bounds__(256) void ln_kernel(
    const float* __restrict__ x, const float* __restrict__ gw,
    const float* __restrict__ bw, __bf16* __restrict__ y)
{
  __shared__ float r1[256], r2[256];
  const long n = blockIdx.x;
  const int c = threadIdx.x * 4;
  const float4 v = *(const float4*)(x + n * D_ + c);
  r1[threadIdx.x] = v.x + v.y + v.z + v.w;
  r2[threadIdx.x] = v.x * v.x + v.y * v.y + v.z * v.z + v.w * v.w;
  __syncthreads();
  for (int off = 128; off; off >>= 1) {
    if (threadIdx.x < off) {
      r1[threadIdx.x] += r1[threadIdx.x + off];
      r2[threadIdx.x] += r2[threadIdx.x + off];
    }
    __syncthreads();
  }
  const float mu = r1[0] * (1.0f / D_);
  const float var = r2[0] * (1.0f / D_) - mu * mu;
  const float rstd = rsqrtf(var + 1e-5f);
  const float4 gv = *(const float4*)(gw + c);
  const float4 bv = *(const float4*)(bw + c);
  __bf16* yr = y + n * D_ + c;
  yr[0] = (__bf16)((v.x - mu) * rstd * gv.x + bv.x);
  yr[1] = (__bf16)((v.y - mu) * rstd * gv.y + bv.y);
  yr[2] = (__bf16)((v.z - mu) * rstd * gv.z + bv.z);
  yr[3] = (__bf16)((v.w - mu) * rstd * gv.w + bv.w);
}

// fp32 -> bf16 straight convert (one float4 group per thread)
__global__ __launch_bounds__(256) void cvt_kernel(
    const float* __restrict__ in, __bf16* __restrict__ out, long n4)
{
  const long i = (long)blockIdx.x * 256 + threadIdx.x;
  if (i >= n4) return;
  const float4 v = *(const float4*)(in + i * 4);
  __bf16* o = out + i * 4;
  o[0] = (__bf16)v.x; o[1] = (__bf16)v.y; o[2] = (__bf16)v.z; o[3] = (__bf16)v.w;
}

// fp32 [R,C] -> bf16 [C,R] transpose-convert (32x32 LDS tiles, batched)
__global__ __launch_bounds__(256) void cvt_t_kernel(
    const float* __restrict__ in, __bf16* __restrict__ outp,
    int R, int C, long sIn, long sOut)
{
  __shared__ float tile[32][33];
  in   += (long)blockIdx.z * sIn;
  outp += (long)blockIdx.z * sOut;
  const int r0 = blockIdx.y * 32, c0 = blockIdx.x * 32;
  const int tr = threadIdx.x >> 5, tc = threadIdx.x & 31;
#pragma unroll
  for (int i = 0; i < 4; ++i)
    tile[tr + 8 * i][tc] = in[(long)(r0 + tr + 8 * i) * C + c0 + tc];
  __syncthreads();
#pragma unroll
  for (int i = 0; i < 4; ++i)
    outp[(long)(c0 + tr + 8 * i) * R + r0 + tc] = (__bf16)tile[tc][tr + 8 * i];
}

// ---------------------------------------------------------------------------
// Router: logits = x @ router_w[D,E]; softmax/argmax/gate/lse per token.
// One wave per token (8 tokens per 256-thread block). x is bf16.
// ---------------------------------------------------------------------------
__global__ __launch_bounds__(256) void router_kernel(
    const __bf16* __restrict__ x, const float* __restrict__ w,
    float* __restrict__ probs, float* __restrict__ gate,
    int* __restrict__ eidx, float* __restrict__ lsev)
{
  const int token = blockIdx.x * 8 + (threadIdx.x >> 5);
  const int lane = threadIdx.x & 31;
  const __bf16* xr = x + (long)token * D_;
  float a[8] = {};
  for (int d = lane; d < D_; d += 32) {
    const float xv = (float)xr[d];
#pragma unroll
    for (int e = 0; e < 8; ++e) a[e] += xv * w[d * 8 + e];
  }
#pragma unroll
  for (int off = 16; off; off >>= 1)
#pragma unroll
    for (int e = 0; e < 8; ++e) a[e] += __shfl_xor(a[e], off, 32);
  if (lane == 0) {
    float mx = a[0]; int am = 0;
#pragma unroll
    for (int e = 1; e < 8; ++e) if (a[e] > mx) { mx = a[e]; am = e; }
    float sum = 0.0f, p[8];
#pragma unroll
    for (int e = 0; e < 8; ++e) { p[e] = __expf(a[e] - mx); sum += p[e]; }
    const float inv = 1.0f / sum;
#pragma unroll
    for (int e = 0; e < 8; ++e) probs[(long)token * 8 + e] = p[e] * inv;
    gate[token] = p[am] * inv;
    eidx[token] = am;
    lsev[token] = mx + __logf(sum);
  }
}

// Deterministic per-expert queue positions (matches reference cumsum order).
__global__ void pos_kernel(const int* __restrict__ eidx,
                           const unsigned char* __restrict__ tmask,
                           int* __restrict__ pos, int* __restrict__ keep)
{
  if (threadIdx.x != 0 || blockIdx.x != 0) return;
  int cnt[8];
  for (int e = 0; e < 8; ++e) cnt[e] = 0;
  for (int n = 0; n < N_; ++n) {
    if (tmask[n]) {
      const int e = eidx[n];
      const int p = cnt[e]++;
      pos[n]  = (p < CAP_) ? p : (CAP_ - 1);
      keep[n] = (p < CAP_) ? 1 : 0;
    } else { pos[n] = 0; keep[n] = 0; }
  }
}

// Scatter kept tokens (bf16) into expert buffers. 2048B per token.
__global__ __launch_bounds__(256) void dispatch_kernel(
    const __bf16* __restrict__ x, const int* __restrict__ eidx,
    const int* __restrict__ pos, const int* __restrict__ keep,
    __bf16* __restrict__ buf)
{
  const int n = blockIdx.x;
  if (!keep[n]) return;
  const unsigned long long* src =
      (const unsigned long long*)(x + (long)n * D_);
  unsigned long long* dst =
      (unsigned long long*)(buf + ((long)eidx[n] * CAP_ + pos[n]) * D_);
  dst[threadIdx.x] = src[threadIdx.x];
}

__global__ __launch_bounds__(256) void combine_kernel(
    const float* __restrict__ src2, const float* __restrict__ o,
    const int* __restrict__ eidx, const int* __restrict__ pos,
    const int* __restrict__ keep, const float* __restrict__ gate,
    float* __restrict__ out)
{
  const long n = blockIdx.x;
  const int c = threadIdx.x * 4;
  float4 base = *(const float4*)(src2 + n * D_ + c);
  if (keep[n]) {
    const float gv = gate[n];
    const float4 ov = *(const float4*)(o + ((long)eidx[n] * CAP_ + pos[n]) * D_ + c);
    base.x += gv * ov.x; base.y += gv * ov.y;
    base.z += gv * ov.z; base.w += gv * ov.w;
  }
  *(float4*)(out + n * D_ + c) = base;
}

// lb_loss = E * sum_e(frac_e * pmean_e); z_loss = sum(lse^2 * mask)/ntok
__global__ __launch_bounds__(256) void loss_kernel(
    const float* __restrict__ probs, const int* __restrict__ eidx,
    const float* __restrict__ lsev, const unsigned char* __restrict__ tmask,
    float* __restrict__ out2)
{
  __shared__ float red[256];
  __shared__ float tot[18];
  float loc[18];
  for (int q = 0; q < 18; ++q) loc[q] = 0.0f;
  for (int n = threadIdx.x; n < N_; n += 256) {
    if (tmask[n]) {
      loc[17] += 1.0f;
      loc[8 + eidx[n]] += 1.0f;
      loc[16] += lsev[n] * lsev[n];
#pragma unroll
      for (int e = 0; e < 8; ++e) loc[e] += probs[(long)n * 8 + e];
    }
  }
  for (int q = 0; q < 18; ++q) {
    red[threadIdx.x] = loc[q];
    __syncthreads();
    for (int off = 128; off; off >>= 1) {
      if (threadIdx.x < off) red[threadIdx.x] += red[threadIdx.x + off];
      __syncthreads();
    }
    if (threadIdx.x == 0) tot[q] = red[0];
    __syncthreads();
  }
  if (threadIdx.x == 0) {
    const float ntok = fmaxf(tot[17], 1.0f);
    float lb = 0.0f;
    for (int e = 0; e < 8; ++e) lb += (tot[8 + e] / ntok) * (tot[e] / ntok);
    out2[0] = 8.0f * lb;
    out2[1] = tot[16] / ntok;
  }
}

// ---------------------------------------------------------------------------
extern "C" void kernel_launch(void* const* d_in, const int* in_sizes, int n_in,
                              void* d_out, int out_size, void* d_ws, size_t ws_size,
                              hipStream_t stream)
{
  const float* src           = (const float*)d_in[0];
  const unsigned char* pad   = (const unsigned char*)d_in[1];
  const unsigned char* tmask = (const unsigned char*)d_in[2];
  const float* ln1_g = (const float*)d_in[3];
  const float* ln1_b = (const float*)d_in[4];
  const float* w_qkv = (const float*)d_in[5];
  const float* b_qkv = (const float*)d_in[6];
  const float* w_o   = (const float*)d_in[7];
  const float* b_o   = (const float*)d_in[8];
  const float* ln2_g = (const float*)d_in[9];
  const float* ln2_b = (const float*)d_in[10];
  const float* w_r   = (const float*)d_in[11];
  const float* w1    = (const float*)d_in[12];
  const float* b1    = (const float*)d_in[13];
  const float* w2    = (const float*)d_in[14];
  const float* b2    = (const float*)d_in[15];
  float* out = (float*)d_out;
  (void)in_sizes; (void)n_in; (void)out_size; (void)ws_size;

  char* ws = (char*)d_ws;
  size_t off = 0;
  auto alloc = [&](size_t bytes) {
    void* p = ws + off;
    off += (bytes + 255) & ~size_t(255);
    return p;
  };
  __bf16* xln_bf  = (__bf16*)alloc((size_t)N_ * D_ * 2);
  float*  qkv     = (float*)alloc((size_t)N_ * TD_ * 4);
  __bf16* ctx_bf  = (__bf16*)alloc((size_t)N_ * D_ * 2);
  float*  src2    = (float*)alloc((size_t)N_ * D_ * 4);
  __bf16* xln2_bf = (__bf16*)alloc((size_t)N_ * D_ * 2);
  float*  probs   = (float*)alloc((size_t)N_ * 8 * 4);
  float*  gate    = (float*)alloc((size_t)N_ * 4);
  float*  lsev    = (float*)alloc((size_t)N_ * 4);
  int*    eidx    = (int*)alloc((size_t)N_ * 4);
  int*    pos     = (int*)alloc((size_t)N_ * 4);
  int*    keep    = (int*)alloc((size_t)N_ * 4);
  __bf16* buf_bf  = (__bf16*)alloc((size_t)E_ * CAP_ * D_ * 2);
  __bf16* hbuf_bf = (__bf16*)alloc((size_t)E_ * CAP_ * DFF_ * 2);
  float*  obuf    = (float*)alloc((size_t)E_ * CAP_ * D_ * 4);
  __bf16* wqkv_bf = (__bf16*)alloc((size_t)TD_ * D_ * 2);
  __bf16* wo_bf   = (__bf16*)alloc((size_t)D_ * D_ * 2);
  __bf16* w1t_bf  = (__bf16*)alloc((size_t)E_ * D_ * DFF_ * 2);  // [E][DFF,D]
  __bf16* w2t_bf  = (__bf16*)alloc((size_t)E_ * DFF_ * D_ * 2);  // [E][D,DFF]

  // 0. Weight conversion (bf16, and transpose MoE weights to K-contiguous)
  cvt_kernel<<<(TD_ * D_ / 4 + 255) / 256, 256, 0, stream>>>(
      w_qkv, wqkv_bf, (long)TD_ * D_ / 4);
  cvt_kernel<<<(D_ * D_ / 4 + 255) / 256, 256, 0, stream>>>(
      w_o, wo_bf, (long)D_ * D_ / 4);
  cvt_t_kernel<<<dim3(DFF_ / 32, D_ / 32, E_), 256, 0, stream>>>(
      w1, w1t_bf, D_, DFF_, (long)D_ * DFF_, (long)D_ * DFF_);
  cvt_t_kernel<<<dim3(D_ / 32, DFF_ / 32, E_), 256, 0, stream>>>(
      w2, w2t_bf, DFF_, D_, (long)DFF_ * D_, (long)DFF_ * D_);

  // 1. LN1 -> bf16
  ln_kernel<<<N_, 256, 0, stream>>>(src, ln1_g, ln1_b, xln_bf);
  // 2. QKV projection: xln[N,D] @ in_proj_w^T -> qkv fp32 (NT)
  gemm_wmma_nt<<<dim3(TD_ / 128, N_ / 128, 1), 256, 0, stream>>>(
      xln_bf, wqkv_bf, qkv, b_qkv, nullptr, N_, TD_, D_, 0, 0, 0, 0, 0, 0);
  // 3. Attention -> bf16 ctx
  flash_attn<<<dim3(S_ / 128, B_ * H_), 256, 0, stream>>>(qkv, pad, ctx_bf);
  // 4. Out projection + bias + residual(src) -> src2 fp32 (NT)
  gemm_wmma_nt<<<dim3(D_ / 128, N_ / 128, 1), 256, 0, stream>>>(
      ctx_bf, wo_bf, src2, b_o, src, N_, D_, D_, 0, 0, 0, 0, 0, 0);
  // 5. LN2 -> bf16
  ln_kernel<<<N_, 256, 0, stream>>>(src2, ln2_g, ln2_b, xln2_bf);
  // 6. Router
  router_kernel<<<N_ / 8, 256, 0, stream>>>(xln2_bf, w_r, probs, gate, eidx, lsev);
  // 7. Deterministic capacity positions
  pos_kernel<<<1, 32, 0, stream>>>(eidx, tmask, pos, keep);
  // 8. Zero dispatch buffer, scatter kept tokens
  (void)hipMemsetAsync(buf_bf, 0, (size_t)E_ * CAP_ * D_ * 2, stream);
  dispatch_kernel<<<N_, 256, 0, stream>>>(xln2_bf, eidx, pos, keep, buf_bf);
  // 9. Expert FFN: h = relu(buf @ w1 + b1) -> bf16 (NT vs transposed w1)
  gemm_wmma_nt<<<dim3(DFF_ / 128, CAP_ / 128, E_), 256, 0, stream>>>(
      buf_bf, w1t_bf, hbuf_bf, b1, nullptr, CAP_, DFF_, D_,
      (long)CAP_ * D_, (long)D_ * DFF_, (long)CAP_ * DFF_, DFF_, 1, 1);
  //10. o = h @ w2 + b2 -> fp32 (NT vs transposed w2)
  gemm_wmma_nt<<<dim3(D_ / 128, CAP_ / 128, E_), 256, 0, stream>>>(
      hbuf_bf, w2t_bf, obuf, b2, nullptr, CAP_, D_, DFF_,
      (long)CAP_ * DFF_, (long)DFF_ * D_, (long)CAP_ * D_, D_, 0, 0);
  //11. Combine + final residual -> d_out
  combine_kernel<<<N_, 256, 0, stream>>>(src2, obuf, eidx, pos, keep, gate, out);
  //12. Aux losses -> d_out[N*D], d_out[N*D+1]
  loss_kernel<<<1, 256, 0, stream>>>(probs, eidx, lsev, tmask, out + (size_t)N_ * D_);
}
